// interpolater_61452392071435
// MI455X (gfx1250) — matrix operand order repaired and checked by reading, not compiled, not run on hardware
//
#include <hip/hip_runtime.h>

#define Bsz 128
#define Cch 64
#define Hh  32
#define Tt  256

typedef __attribute__((ext_vector_type(16))) __bf16 v16bf;
typedef __attribute__((ext_vector_type(8)))  float  v8f;

// h-state row stride in bf16 elements: 40 halfs = 80 bytes (16B aligned, spreads LDS banks)
#define SH_STRIDE 40

// CDNA5 async DMA: global -> LDS, tracked on ASYNCcnt (no VGPR round trip).
__device__ __forceinline__ void async_f32_to_lds(unsigned lds_off, const float* g) {
    asm volatile("global_load_async_to_lds_b32 %0, %1, off"
                 :: "v"(lds_off), "v"((unsigned long long)(size_t)g)
                 : "memory");
}
__device__ __forceinline__ void wait_asynccnt0() {
    asm volatile("s_wait_asynccnt 0x0" ::: "memory");
}

__device__ __forceinline__ float bf16lo(unsigned w) { return __uint_as_float(w << 16); }
__device__ __forceinline__ float bf16hi(unsigned w) { return __uint_as_float(w & 0xffff0000u); }
__device__ __forceinline__ float relu1(float v) {
    return __builtin_amdgcn_fmed3f(v, 0.0f, 3.402823466e+38f);  // single v_med3_f32
}

__global__ __launch_bounds__(512)
void mrnn_scan_kernel(const float* __restrict__ x,
                      const float* __restrict__ mi,
                      const float* __restrict__ di,
                      const float* __restrict__ Wf, const float* __restrict__ Vf, const float* __restrict__ cf,
                      const float* __restrict__ Wb, const float* __restrict__ Vb, const float* __restrict__ cb,
                      const float* __restrict__ U,
                      float* __restrict__ partial)  // [2][B][C][T]
{
    const int blk = blockIdx.x;      // 0..127
    const int c   = blk >> 1;        // channel
    const int dir = blk & 1;         // 0 = forward, 1 = backward

    const int tid   = threadIdx.x;
    const int lane  = tid & 31;      // wave32
    const int wave  = tid >> 5;      // 0..15
    const int mTile = wave >> 1;     // 0..7
    const int nTile = wave & 1;      // 0..1

    const float* W  = dir ? Wb : Wf;
    const float* V  = dir ? Vb : Vf;
    const float* cv = dir ? cb : cf;

    __shared__ __align__(16) __bf16 sH[2][Bsz * SH_STRIDE];  // double-buffered h state, 20 KB
    __shared__ __align__(16) float  sFeat[2][3][Bsz];        // double-buffered per-step inputs

    // ---------------- preload B fragment: B[k][n] = W[c][n][k]  (32x16 bf16) ----------------
    const int nLoc  = lane & 15;
    const int nGlob = nTile * 16 + nLoc;        // output column
    const int kBase = (lane >> 4) * 8;
    v16bf bfrag;
    {
        const float* wrow = W + (size_t)c * Hh * Hh + (size_t)nGlob * Hh + kBase;
        #pragma unroll
        for (int j = 0; j < 16; ++j) bfrag[j] = (__bf16)wrow[j + (j & 8)];
    }
    // per-lane input-projection constants for column nGlob
    const float v0c = V[((size_t)c * Hh + nGlob) * 3 + 0];
    const float v1c = V[((size_t)c * Hh + nGlob) * 3 + 1];
    const float v2c = V[((size_t)c * Hh + nGlob) * 3 + 2];
    const float bC  = cv[(size_t)c * Hh + nGlob];

    // U slice for this direction: uniform address -> scalar loads (SGPRs)
    float ureg[Hh];
    #pragma unroll
    for (int n = 0; n < Hh; ++n)
        ureg[n] = U[(size_t)c * (2 * Hh) + (size_t)dir * Hh + n];

    // zero initial h state (buffer 0 only; buffer 1 fully written each step)
    for (int i = tid; i < (Bsz * SH_STRIDE) / 2; i += 512)
        ((unsigned int*)sH[0])[i] = 0u;

    // async preload features for t = 0 into sFeat[0]
    if (tid < 3 * Bsz) {
        const int t_in0 = dir ? (Tt - 1) : 0;                 // edge padding
        const int j = tid >> 7;                               // 0:x 1:m 2:d
        const int b = tid & 127;
        const float* src = (j == 0) ? x : (j == 1 ? mi : di);
        async_f32_to_lds((unsigned)(size_t)&sFeat[0][j][b],
                         src + (size_t)b * Cch * Tt + (size_t)c * Tt + t_in0);
    }
    wait_asynccnt0();
    __syncthreads();

    float* myPartial = partial + (size_t)dir * Bsz * Cch * Tt
                     + (tid < Bsz ? ((size_t)tid * Cch * Tt + (size_t)c * Tt) : 0);

    const int rb = mTile * 16 + (lane >> 4) * 8;   // first C/D row for this lane

    for (int t = 0; t < Tt; ++t) {
        const int cur = t & 1;
        const int nxt = cur ^ 1;
        const __bf16* sHc = sH[cur];
        __bf16*       sHn = sH[nxt];

        // ---- A fragment from sH[cur] (two 16B chunks per lane) ----
        v16bf afrag;
        {
            const int r  = mTile * 16 + (lane & 15);
            const int hb = lane >> 4;
            const __bf16* rowp = sHc + r * SH_STRIDE;
            union { v16bf v; uint4 q[2]; } au;
            au.q[0] = *(const uint4*)(rowp + hb * 8);
            au.q[1] = *(const uint4*)(rowp + 16 + hb * 8);
            afrag = au.v;
        }

        // ---- C accumulator = V[c] @ feats + bias (vectorized LDS reads) ----
        v8f acc;
        {
            union { float4 v[2]; float f[8]; } fx, fm, fd;
            fx.v[0] = *(const float4*)&sFeat[cur][0][rb];
            fx.v[1] = *(const float4*)&sFeat[cur][0][rb + 4];
            fm.v[0] = *(const float4*)&sFeat[cur][1][rb];
            fm.v[1] = *(const float4*)&sFeat[cur][1][rb + 4];
            fd.v[0] = *(const float4*)&sFeat[cur][2][rb];
            fd.v[1] = *(const float4*)&sFeat[cur][2][rb + 4];
            #pragma unroll
            for (int i = 0; i < 8; ++i)
                acc[i] = fmaf(v0c, fx.f[i], fmaf(v1c, fm.f[i], fmaf(v2c, fd.f[i], bC)));
        }

        // ---- h_next(tile) = relu( h_t x W^T + acc ) ----
        v8f dacc = __builtin_amdgcn_wmma_f32_16x16x32_bf16(
                       false, afrag, false, bfrag, (short)0, acc, false, false);
        #pragma unroll
        for (int i = 0; i < 8; ++i) dacc[i] = relu1(dacc[i]);

        // ---- store h_{t+1} into sH[nxt] ----
        #pragma unroll
        for (int i = 0; i < 8; ++i)
            sHn[(rb + i) * SH_STRIDE + nGlob] = (__bf16)dacc[i];

        // ---- output partial for step t-1 from sH[cur] (vectorized bf16 row read) ----
        if (t > 0 && tid < Bsz) {
            const uint4* rowq = (const uint4*)(sHc + tid * SH_STRIDE);
            float s = 0.0f;
            #pragma unroll
            for (int q = 0; q < 4; ++q) {
                const uint4 w = rowq[q];
                const int n0 = q * 8;
                s = fmaf(ureg[n0 + 0], bf16lo(w.x), s);
                s = fmaf(ureg[n0 + 1], bf16hi(w.x), s);
                s = fmaf(ureg[n0 + 2], bf16lo(w.y), s);
                s = fmaf(ureg[n0 + 3], bf16hi(w.y), s);
                s = fmaf(ureg[n0 + 4], bf16lo(w.z), s);
                s = fmaf(ureg[n0 + 5], bf16hi(w.z), s);
                s = fmaf(ureg[n0 + 6], bf16lo(w.w), s);
                s = fmaf(ureg[n0 + 7], bf16hi(w.w), s);
            }
            const int tprev = t - 1;
            const int t_out = dir ? (Tt - 1 - tprev) : tprev;
            myPartial[t_out] = s;
        }

        // ---- async prefetch features for t+1 into sFeat[nxt] ----
        if (t + 1 < Tt && tid < 3 * Bsz) {
            const int tn  = t + 1;
            const int t_b = Tt - tn;
            const int t_in = dir ? (t_b < (Tt - 1) ? t_b : (Tt - 1)) : (tn - 1);
            const int j = tid >> 7;
            const int b = tid & 127;
            const float* src = (j == 0) ? x : (j == 1 ? mi : di);
            async_f32_to_lds((unsigned)(size_t)&sFeat[nxt][j][b],
                             src + (size_t)b * Cch * Tt + (size_t)c * Tt + t_in);
        }

        wait_asynccnt0();   // DMA has had the whole step to complete
        __syncthreads();    // single barrier per step (double-buffered sH)
    }

    // final output partial for time T-1 (state lives in sH[Tt & 1] == sH[0])
    if (tid < Bsz) {
        const uint4* rowq = (const uint4*)(sH[Tt & 1] + tid * SH_STRIDE);
        float s = 0.0f;
        #pragma unroll
        for (int q = 0; q < 4; ++q) {
            const uint4 w = rowq[q];
            const int n0 = q * 8;
            s = fmaf(ureg[n0 + 0], bf16lo(w.x), s);
            s = fmaf(ureg[n0 + 1], bf16hi(w.x), s);
            s = fmaf(ureg[n0 + 2], bf16lo(w.y), s);
            s = fmaf(ureg[n0 + 3], bf16hi(w.y), s);
            s = fmaf(ureg[n0 + 4], bf16lo(w.z), s);
            s = fmaf(ureg[n0 + 5], bf16hi(w.z), s);
            s = fmaf(ureg[n0 + 6], bf16lo(w.w), s);
            s = fmaf(ureg[n0 + 7], bf16hi(w.w), s);
        }
        const int t_out = dir ? 0 : (Tt - 1);
        myPartial[t_out] = s;
    }
}

__global__ __launch_bounds__(256)
void mrnn_combine_kernel(const float* __restrict__ partial,
                         const float* __restrict__ c0,
                         float* __restrict__ out, int n)
{
    const int i = blockIdx.x * blockDim.x + threadIdx.x;
    if (i >= n) return;
    const int c = (i / Tt) % Cch;
    const float v = partial[i] + partial[(size_t)n + i] + c0[c];
    out[i] = __builtin_amdgcn_fmed3f(v, 0.0f, 3.402823466e+38f);
}

extern "C" void kernel_launch(void* const* d_in, const int* in_sizes, int n_in,
                              void* d_out, int out_size, void* d_ws, size_t ws_size,
                              hipStream_t stream)
{
    const float* x  = (const float*)d_in[0];
    const float* m  = (const float*)d_in[1];
    const float* d  = (const float*)d_in[2];
    const float* Wf = (const float*)d_in[3];
    const float* Vf = (const float*)d_in[4];
    const float* cf = (const float*)d_in[5];
    const float* Wb = (const float*)d_in[6];
    const float* Vb = (const float*)d_in[7];
    const float* cb = (const float*)d_in[8];
    const float* U  = (const float*)d_in[9];
    const float* c0 = (const float*)d_in[10];

    float* partial = (float*)d_ws;           // 2 * B*C*T floats = 16 MB
    float* out     = (float*)d_out;

    mrnn_scan_kernel<<<Cch * 2, 512, 0, stream>>>(x, m, d, Wf, Vf, cf, Wb, Vb, cb, U, partial);

    const int n = Bsz * Cch * Tt;
    mrnn_combine_kernel<<<(n + 255) / 256, 256, 0, stream>>>(partial, c0, out, n);
}
